// UNet_26285199852085
// MI455X (gfx1250) — matrix operand order repaired and checked by reading, not compiled, not run on hardware
//
#include <hip/hip_runtime.h>
#include <math.h>

typedef __attribute__((ext_vector_type(16))) _Float16 v16h;
typedef __attribute__((ext_vector_type(8)))  _Float16 h8v;
typedef __attribute__((ext_vector_type(8)))  float    v8f;
typedef __attribute__((ext_vector_type(4)))  float    f32x4;

#define ACT_NONE 0
#define ACT_SOFTPLUS 1
#define TK 32

__device__ __forceinline__ float sigmoidf_(float x) { return 1.f / (1.f + __expf(-x)); }

// ---------------------------------------------------------------------------
// GEMM: C[M,N] = act(A[M,K] @ W[N,K]^T + bias), fp32 in/out, f16 WMMA compute.
// Workgroup = 4 waves, 64x64 output tile. Shared 64x32 B tile staged in LDS
// (f16, double-buffered, one barrier/iteration); each wave: 1 A-frag (regs)
// feeds 4 WMMAs per K-step. Caller guarantees M % 64 == 0, K % 4 == 0,
// lda % 4 == 0, 16B-aligned pointers. Ragged N: clamped rows + store guard.
// Ragged K: branch-free masked register tail.
// ---------------------------------------------------------------------------
__global__ void gemm_f16wmma_kernel(const float* __restrict__ A, int lda,
                                    const float* __restrict__ W,
                                    const float* __restrict__ bias,
                                    float* __restrict__ C, int ldc,
                                    int M, int N, int K, int act)
{
    __shared__ _Float16 Bs[2][64][TK];   // 8 KB

    int tid  = threadIdx.x;
    int lane = tid & 31;
    int wave = tid >> 5;
    int n0 = blockIdx.x * 64;
    int m0 = blockIdx.y * 64 + wave * 16;
    int l15 = lane & 15;
    int hi  = lane >> 4;           // half-wave select
    int khA = hi ? 8 : 0;
    int khB = hi ? 8 : 0;

    const float* arow = A + (size_t)(m0 + l15) * (size_t)lda;

    // cooperative B staging coords: thread -> (row 0..63, col 0 or 16)
    int srow = tid >> 1;
    int scol = (tid & 1) * 16;
    int grow = n0 + srow; if (grow > N - 1) grow = N - 1;
    const float* wstage = W + (size_t)grow * (size_t)K + scol;

    v8f acc[4] = {};
    int Kfull = K & ~(TK - 1);

    int buf = 0;
    for (int k0 = 0; k0 < Kfull; k0 += TK, buf ^= 1) {
        // ---- stage B tile (global f32 -> LDS f16) ----
        {
            const f32x4* p = (const f32x4*)(wstage + k0);
            f32x4 w0 = p[0], w1 = p[1], w2 = p[2], w3 = p[3];
            h8v x0, x1;
#pragma unroll
            for (int e = 0; e < 4; ++e) {
                x0[e]     = (_Float16)w0[e];
                x0[4 + e] = (_Float16)w1[e];
                x1[e]     = (_Float16)w2[e];
                x1[4 + e] = (_Float16)w3[e];
            }
            _Float16* dst = &Bs[buf][srow][scol];
            *(h8v*)(dst)     = x0;
            *(h8v*)(dst + 8) = x1;
        }
        __syncthreads();

        // ---- A fragment: K = k0+khA+{0..7} and k0+khA+{16..23} ----
        v16h a;
        {
            const f32x4* pa = (const f32x4*)(arow + k0 + khA);
            f32x4 a0 = pa[0], a1 = pa[1], a2 = pa[4], a3 = pa[5];
#pragma unroll
            for (int e = 0; e < 4; ++e) {
                a[e]      = (_Float16)a0[e];
                a[4 + e]  = (_Float16)a1[e];
                a[8 + e]  = (_Float16)a2[e];
                a[12 + e] = (_Float16)a3[e];
            }
        }

        // ---- B fragments from LDS, 4 WMMAs ----
#pragma unroll
        for (int j = 0; j < 4; ++j) {
            const h8v* pb = (const h8v*)&Bs[buf][j * 16 + l15][khB];
            h8v lo = pb[0], hiv = pb[2];
            v16h b = __builtin_shufflevector(lo, hiv,
                                             0, 1, 2, 3, 4, 5, 6, 7,
                                             8, 9, 10, 11, 12, 13, 14, 15);
            acc[j] = __builtin_amdgcn_wmma_f32_16x16x32_f16(false, a, false, b,
                                                            (short)0, acc[j], false, false);
        }
    }

    if (Kfull < K) {
        // Masked tail: clamped-index unconditional loads + selects (no branches).
        const float* wrow[4];
#pragma unroll
        for (int j = 0; j < 4; ++j) {
            int nc = n0 + j * 16 + l15;
            int nr = (nc < N) ? nc : (N - 1);
            wrow[j] = W + (size_t)nr * (size_t)K;
        }
        v16h a, b[4];
#pragma unroll
        for (int e = 0; e < 16; ++e) {
            int ka = Kfull + khA + e + ((e >= 8) ? 8 : 0);
            int kac = (ka < K) ? ka : (K - 1);
            float av = arow[kac];
            a[e] = (ka < K) ? (_Float16)av : (_Float16)0.f;
            int kb = Kfull + khB + e + ((e >= 8) ? 8 : 0);
            int kbc = (kb < K) ? kb : (K - 1);
#pragma unroll
            for (int j = 0; j < 4; ++j) {
                float wv = wrow[j][kbc];
                b[j][e] = (kb < K) ? (_Float16)wv : (_Float16)0.f;
            }
        }
        acc[0] = __builtin_amdgcn_wmma_f32_16x16x32_f16(false, a, false, b[0], (short)0, acc[0], false, false);
        acc[1] = __builtin_amdgcn_wmma_f32_16x16x32_f16(false, a, false, b[1], (short)0, acc[1], false, false);
        acc[2] = __builtin_amdgcn_wmma_f32_16x16x32_f16(false, a, false, b[2], (short)0, acc[2], false, false);
        acc[3] = __builtin_amdgcn_wmma_f32_16x16x32_f16(false, a, false, b[3], (short)0, acc[3], false, false);
    }

#pragma unroll
    for (int j = 0; j < 4; ++j) {
        int col = n0 + j * 16 + l15;
        if (col >= N) continue;
        float bv = bias ? bias[col] : 0.f;
#pragma unroll
        for (int r = 0; r < 8; ++r) {
            int row = m0 + r + (hi ? 8 : 0);
            float v = acc[j][r] + bv;
            if (act == ACT_SOFTPLUS) v = (v > 20.f) ? v : log1pf(__expf(v));
            C[(size_t)row * (size_t)ldc + col] = v;
        }
    }
}

// ---------------------------------------------------------------------------
// PatchEmbed: 2x2 conv stride 2 on 1x256x256 + bias + positional embedding.
// ---------------------------------------------------------------------------
__global__ void patch_embed_kernel(const float* __restrict__ inp,
                                   const float* __restrict__ pw,
                                   const float* __restrict__ pb,
                                   const float* __restrict__ pos,
                                   float* __restrict__ out)
{
    int idx = blockIdx.x * blockDim.x + threadIdx.x;
    if (idx >= 16384 * 64) return;
    int o = idx & 63;
    int l = idx >> 6;
    int h = l >> 7, w = l & 127;
    const float* base = inp + (size_t)(2 * h) * 256 + 2 * w;
    float acc = pb[o];
    acc += base[0]   * pw[o * 4 + 0];
    acc += base[1]   * pw[o * 4 + 1];
    acc += base[256] * pw[o * 4 + 2];
    acc += base[257] * pw[o * 4 + 3];
    out[idx] = acc + pos[idx];
}

// Causal depthwise conv (kernel 4) + bias + SiLU.  xi = xz[:, :di] (row stride ldx)
__global__ void dwconv_silu_kernel(const float* __restrict__ xz, int ldx,
                                   const float* __restrict__ cw,
                                   const float* __restrict__ cb,
                                   float* __restrict__ xc, int L, int di)
{
    int idx = blockIdx.x * blockDim.x + threadIdx.x;
    if (idx >= L * di) return;
    int d = idx % di;
    int l = idx / di;
    float acc = cb[d];
#pragma unroll
    for (int k = 0; k < 4; ++k) {
        int t = l - 3 + k;
        if (t >= 0) acc += xz[(size_t)t * ldx + d] * cw[k * di + d];
    }
    xc[idx] = acc * sigmoidf_(acc);
}

// ---------------------------------------------------------------------------
// Selective scan: one thread per channel d, 16 states in registers,
// sequential over L.  xdbl rows: [dt(dtr) | B(16) | C(16)], dtr % 4 == 0.
// ---------------------------------------------------------------------------
__global__ void scan_kernel(const float* __restrict__ dt,
                            const float* __restrict__ xc,
                            const float* __restrict__ xdbl, int ldx, int dtr,
                            const float* __restrict__ A_log,
                            const float* __restrict__ Dp,
                            float* __restrict__ y, int L, int di)
{
    int d = blockIdx.x * blockDim.x + threadIdx.x;
    if (d >= di) return;
    float a[16], h[16];
#pragma unroll
    for (int n = 0; n < 16; ++n) {
        a[n] = -__expf(A_log[d * 16 + n]);
        h[n] = 0.f;
    }
    float Dv = Dp[d];
    for (int t = 0; t < L; ++t) {
        float dtv = dt[(size_t)t * di + d];
        float xv  = xc[(size_t)t * di + d];
        const f32x4* rp = (const f32x4*)(xdbl + (size_t)t * ldx + dtr);
        f32x4 Bv[4], Cv[4];
#pragma unroll
        for (int q = 0; q < 4; ++q) { Bv[q] = rp[q]; Cv[q] = rp[4 + q]; }
        float acc = 0.f;
#pragma unroll
        for (int n = 0; n < 16; ++n) {
            float bn = Bv[n >> 2][n & 3];
            float cn = Cv[n >> 2][n & 3];
            h[n] = __expf(dtv * a[n]) * h[n] + dtv * bn * xv;
            acc += h[n] * cn;
        }
        y[(size_t)t * di + d] = acc + xv * Dv;
    }
}

// y *= silu(z), z = xz[:, di:2di]
__global__ void mul_silu_kernel(float* __restrict__ y,
                                const float* __restrict__ xz, int ldx,
                                int di, int total)
{
    int idx = blockIdx.x * blockDim.x + threadIdx.x;
    if (idx >= total) return;
    int d = idx % di;
    int l = idx / di;
    float z = xz[(size_t)l * ldx + di + d];
    y[idx] *= z * sigmoidf_(z);
}

// LayerNorm over channel dim; one wave per token.
__global__ void layernorm_kernel(const float* __restrict__ in,
                                 const float* __restrict__ g,
                                 const float* __restrict__ b,
                                 float* __restrict__ out, int L, int C)
{
    int l = blockIdx.x;
    int lane = threadIdx.x;
    const float* row = in + (size_t)l * C;
    float s = 0.f, ss = 0.f;
    for (int c = lane; c < C; c += 32) { float v = row[c]; s += v; ss += v * v; }
#pragma unroll
    for (int o = 16; o > 0; o >>= 1) {
        s  += __shfl_xor(s,  o, 32);
        ss += __shfl_xor(ss, o, 32);
    }
    float mu = s / (float)C;
    float var = ss / (float)C - mu * mu;
    float r = rsqrtf(var + 1e-5f);
    for (int c = lane; c < C; c += 32)
        out[(size_t)l * C + c] = (row[c] - mu) * r * g[c] + b[c];
}

// 2x2 maxpool on token-major (H,W,C)
__global__ void maxpool_kernel(const float* __restrict__ in, float* __restrict__ out,
                               int Hout, int Wout, int C)
{
    int idx = blockIdx.x * blockDim.x + threadIdx.x;
    int total = Hout * Wout * C;
    if (idx >= total) return;
    int c = idx % C;
    int l = idx / C;
    int ho = l / Wout, wo = l % Wout;
    int Win = 2 * Wout;
    const float* p = in + ((size_t)(2 * ho) * Win + 2 * wo) * C + c;
    size_t rs = (size_t)Win * C;
    float m = fmaxf(fmaxf(p[0], p[C]), fmaxf(p[rs], p[rs + C]));
    out[idx] = m;
}

// out(Ho,Wo, Cs+Cm) = cat(skip(Ho,Wo,Cs), bilinear2x_align_corners(sm(Hs,Ws,Cm)))
__global__ void upcat_kernel(const float* __restrict__ skip, int Cs,
                             const float* __restrict__ sm, int Cm,
                             int Hs, int Ws,
                             float* __restrict__ out, int Ho, int Wo)
{
    int Ct = Cs + Cm;
    int idx = blockIdx.x * blockDim.x + threadIdx.x;
    int total = Ho * Wo * Ct;
    if (idx >= total) return;
    int c = idx % Ct;
    int l = idx / Ct;
    if (c < Cs) { out[idx] = skip[(size_t)l * Cs + c]; return; }
    c -= Cs;
    int h = l / Wo, w = l % Wo;
    float sy = (float)h * (float)(Hs - 1) / (float)(Ho - 1);
    float sx = (float)w * (float)(Ws - 1) / (float)(Wo - 1);
    int y0 = (int)floorf(sy); int y1 = y0 + 1; if (y1 > Hs - 1) y1 = Hs - 1;
    int x0 = (int)floorf(sx); int x1 = x0 + 1; if (x1 > Ws - 1) x1 = Ws - 1;
    float fy = sy - (float)y0, fx = sx - (float)x0;
    float v00 = sm[((size_t)y0 * Ws + x0) * Cm + c];
    float v01 = sm[((size_t)y0 * Ws + x1) * Cm + c];
    float v10 = sm[((size_t)y1 * Ws + x0) * Cm + c];
    float v11 = sm[((size_t)y1 * Ws + x1) * Cm + c];
    float t0 = v00 * (1.f - fy) + v10 * fy;
    float t1 = v01 * (1.f - fy) + v11 * fy;
    out[idx] = t0 * (1.f - fx) + t1 * fx;
}

// (L,C) tokens -> (C,L) NCHW flat
__global__ void tok2chw_kernel(const float* __restrict__ in, float* __restrict__ out,
                               int L, int C)
{
    int idx = blockIdx.x * blockDim.x + threadIdx.x;
    if (idx >= L * C) return;
    int c = idx % C;
    int l = idx / C;
    out[(size_t)c * L + l] = in[idx];
}

// ===========================================================================
// Host orchestration
// ===========================================================================
enum { P_WIN = 0, P_CONVW, P_CONVB, P_WX, P_WDT, P_BDT, P_ALOG, P_D, P_WOUT, P_LNG, P_LNB };

struct BlockP { const float* p[11]; };

static void load_block(void* const* d_in, int& i, BlockP& b)
{
    for (int k = 0; k < 11; ++k) b.p[k] = (const float*)d_in[i++];
}

struct Arena {
    char* base; size_t off, cap;
    float* alloc(size_t nfloats)
    {
        size_t bytes = (nfloats * sizeof(float) + 255) & ~(size_t)255;
        float* p = (float*)(base + off);
        off += bytes;
        return p;
    }
};

static void launch_gemm(const float* A, int lda, const float* W, const float* bias,
                        float* C, int ldc, int M, int N, int K, int act, hipStream_t s)
{
    dim3 grid((N + 63) / 64, M / 64);   // M always a multiple of 64 here
    gemm_f16wmma_kernel<<<grid, 128, 0, s>>>(A, lda, W, bias, C, ldc, M, N, K, act);
}

static void run_mamba(const float* in, float* out, int L, int dm, int oc,
                      const BlockP& P, Arena& ar, hipStream_t s)
{
    int di = 2 * dm;
    int dtr = (dm + 15) / 16;
    int xw = dtr + 32;
    size_t mark = ar.off;
    float* xz = ar.alloc((size_t)L * 2 * di);
    float* xc = ar.alloc((size_t)L * di);
    float* xd = ar.alloc((size_t)L * xw);
    float* dt = ar.alloc((size_t)L * di);
    float* y  = ar.alloc((size_t)L * di);
    float* mo = ar.alloc((size_t)L * oc);

    const int T = 256;
    int n1 = L * di;

    launch_gemm(in, dm, P.p[P_WIN], nullptr, xz, 2 * di, L, 2 * di, dm, ACT_NONE, s);
    dwconv_silu_kernel<<<(n1 + T - 1) / T, T, 0, s>>>(xz, 2 * di, P.p[P_CONVW], P.p[P_CONVB], xc, L, di);
    launch_gemm(xc, di, P.p[P_WX], nullptr, xd, xw, L, xw, di, ACT_NONE, s);
    launch_gemm(xd, xw, P.p[P_WDT], P.p[P_BDT], dt, di, L, di, dtr, ACT_SOFTPLUS, s);
    scan_kernel<<<(di + 127) / 128, 128, 0, s>>>(dt, xc, xd, xw, dtr, P.p[P_ALOG], P.p[P_D], y, L, di);
    mul_silu_kernel<<<(n1 + T - 1) / T, T, 0, s>>>(y, xz, 2 * di, di, n1);
    launch_gemm(y, di, P.p[P_WOUT], nullptr, mo, oc, L, oc, di, ACT_NONE, s);
    layernorm_kernel<<<L, 32, 0, s>>>(mo, P.p[P_LNG], P.p[P_LNB], out, L, oc);

    ar.off = mark;
}

static void run_double(const float* in, float* out, int L, int din, int dout,
                       const BlockP& b1, const BlockP& b2, Arena& ar, hipStream_t s)
{
    run_mamba(in,  out, L, din,  dout, b1, ar, s);   // in read only by first GEMM
    run_mamba(out, out, L, dout, dout, b2, ar, s);   // safe: in consumed before LN writes
}

extern "C" void kernel_launch(void* const* d_in, const int* in_sizes, int n_in,
                              void* d_out, int out_size, void* d_ws, size_t ws_size,
                              hipStream_t stream)
{
    (void)in_sizes; (void)n_in; (void)out_size;
    int i = 0;
    const float* inp     = (const float*)d_in[i++];
    const float* patch_w = (const float*)d_in[i++];
    const float* patch_b = (const float*)d_in[i++];
    const float* pos     = (const float*)d_in[i++];

    BlockP dbl[9][2];   // inc, down1..down4, up1..up4
    for (int k = 0; k < 9; ++k) { load_block(d_in, i, dbl[k][0]); load_block(d_in, i, dbl[k][1]); }
    BlockP oc; load_block(d_in, i, oc);

    Arena ar{ (char*)d_ws, 0, ws_size };
    // persistent feature maps (token-major)
    float* tok0 = ar.alloc(16384 * 64);
    float* x1 = ar.alloc(16384 * 64);
    float* p1 = ar.alloc(4096 * 64);
    float* x2 = ar.alloc(4096 * 128);
    float* p2 = ar.alloc(1024 * 128);
    float* x3 = ar.alloc(1024 * 256);
    float* p3 = ar.alloc(256 * 256);
    float* x4 = ar.alloc(256 * 512);
    float* p4 = ar.alloc(64 * 512);
    float* x5 = ar.alloc(64 * 1024);
    float* c1 = ar.alloc(256 * 1536);
    float* y1 = ar.alloc(256 * 512);
    float* c2 = ar.alloc(1024 * 768);
    float* y2 = ar.alloc(1024 * 256);
    float* c3 = ar.alloc(4096 * 384);
    float* y3 = ar.alloc(4096 * 128);
    float* c4 = ar.alloc(16384 * 192);
    float* y4 = ar.alloc(16384 * 64);
    float* ot = ar.alloc(16384 * 4);

    const int T = 256;
    int n;

    n = 16384 * 64;
    patch_embed_kernel<<<(n + T - 1) / T, T, 0, stream>>>(inp, patch_w, patch_b, pos, tok0);

    run_double(tok0, x1, 16384, 64, 64, dbl[0][0], dbl[0][1], ar, stream);

    n = 4096 * 64;   maxpool_kernel<<<(n + T - 1) / T, T, 0, stream>>>(x1, p1, 64, 64, 64);
    run_double(p1, x2, 4096, 64, 128, dbl[1][0], dbl[1][1], ar, stream);

    n = 1024 * 128;  maxpool_kernel<<<(n + T - 1) / T, T, 0, stream>>>(x2, p2, 32, 32, 128);
    run_double(p2, x3, 1024, 128, 256, dbl[2][0], dbl[2][1], ar, stream);

    n = 256 * 256;   maxpool_kernel<<<(n + T - 1) / T, T, 0, stream>>>(x3, p3, 16, 16, 256);
    run_double(p3, x4, 256, 256, 512, dbl[3][0], dbl[3][1], ar, stream);

    n = 64 * 512;    maxpool_kernel<<<(n + T - 1) / T, T, 0, stream>>>(x4, p4, 8, 8, 512);
    run_double(p4, x5, 64, 512, 1024, dbl[4][0], dbl[4][1], ar, stream);

    n = 256 * 1536;  upcat_kernel<<<(n + T - 1) / T, T, 0, stream>>>(x4, 512, x5, 1024, 8, 8, c1, 16, 16);
    run_double(c1, y1, 256, 1536, 512, dbl[5][0], dbl[5][1], ar, stream);

    n = 1024 * 768;  upcat_kernel<<<(n + T - 1) / T, T, 0, stream>>>(x3, 256, y1, 512, 16, 16, c2, 32, 32);
    run_double(c2, y2, 1024, 768, 256, dbl[6][0], dbl[6][1], ar, stream);

    n = 4096 * 384;  upcat_kernel<<<(n + T - 1) / T, T, 0, stream>>>(x2, 128, y2, 256, 32, 32, c3, 64, 64);
    run_double(c3, y3, 4096, 384, 128, dbl[7][0], dbl[7][1], ar, stream);

    n = 16384 * 192; upcat_kernel<<<(n + T - 1) / T, T, 0, stream>>>(x1, 64, y3, 128, 64, 64, c4, 128, 128);
    run_double(c4, y4, 16384, 192, 64, dbl[8][0], dbl[8][1], ar, stream);

    run_mamba(y4, ot, 16384, 64, 4, oc, ar, stream);

    n = 16384 * 4;
    tok2chw_kernel<<<(n + T - 1) / T, T, 0, stream>>>(ot, (float*)d_out, 16384, 4);
}
